// MinibatchDiscrimination_52939766890699
// MI455X (gfx1250) — compile-verified, matrix operand order checked by reading, not run on hardware
//
#include <hip/hip_runtime.h>
#include <hip/hip_bf16.h>

typedef __attribute__((ext_vector_type(16))) _Float16 v16h;
typedef __attribute__((ext_vector_type(8)))  float    v8f;

#define N_TOTAL        4096   // B*S
#define F_DIM          64
#define KD             512    // K*D
#define K_KER          32
#define D_KER          16
#define OUT_F          96     // F + K
#define ROWS_PER_BLOCK 16
#define THREADS        256
#define T_FRAG_ELEMS   (32 * 2 * 32 * 16)   // tiles * kk * lanes * i = 32768 halfs (64 KB)

// ---------------------------------------------------------------------------
// Prologue: convert T (64x512 f32) once into WMMA-B fragment order, f16.
// Layout: wsT[(((tile*2 + kk)*32 + lane)*16) + i] = (f16) T[k(kk,lane,i)][tile*16 + lane%16]
// with k = kk*32 + (lane>=16)*8 + (i<8 ? i : i+8)  -- the CDNA5 16-bit B layout.
// ---------------------------------------------------------------------------
__global__ __launch_bounds__(THREADS)
void prep_T_kernel(const float* __restrict__ T, _Float16* __restrict__ wsT)
{
    const int idx  = blockIdx.x * THREADS + threadIdx.x;   // 0..32767
    const int i    = idx & 15;
    const int lane = (idx >> 4) & 31;
    const int kk   = (idx >> 9) & 1;
    const int tile = idx >> 10;           // 0..31
    const int half = lane >> 4;
    const int r    = lane & 15;
    const int k    = kk * 32 + half * 8 + (i < 8 ? i : i + 8);
    const int col  = tile * 16 + r;
    wsT[idx] = (_Float16)T[(size_t)k * KD + col];
}

// ---------------------------------------------------------------------------
// Fused main kernel: GEMM tile (WMMA) -> LDS -> minibatch-discrimination feats
// ---------------------------------------------------------------------------
__global__ __launch_bounds__(THREADS)
void mbd_fused_kernel(const float* __restrict__ x,
                      const _Float16* __restrict__ wsT,
                      float* __restrict__ out)
{
    // 16 rows x 512 cols of the intermediate M, fp32 (32 KB of 320 KB WGP LDS).
    __shared__ __align__(16) float Mtile[ROWS_PER_BLOCK * KD];

    const int tid  = threadIdx.x;
    const int lane = tid & 31;
    const int wave = tid >> 5;     // 0..7
    const int half = lane >> 4;    // 0 / 1
    const int r    = lane & 15;    // A: row; B/C/D: column
    const int n0   = blockIdx.x * ROWS_PER_BLOCK;

    // ---- Phase 1: M[16x512] = x[16x64] * T[64x512] via v_wmma_f32_16x16x32_f16 ----
    // A (16-bit, 16x32): lane holds row M = lane%16; per lane K = two contiguous runs
    // of 8 -> load as 4 float4s and pack-convert.
    v16h afrag[2];
    {
        const float4* x4 = (const float4*)(x + (size_t)(n0 + r) * F_DIM);
#pragma unroll
        for (int kk = 0; kk < 2; ++kk) {
            const int q = (kk * 32 + half * 8) >> 2;   // float4 index of first run
            const float4 c0 = x4[q + 0];
            const float4 c1 = x4[q + 1];
            const float4 c2 = x4[q + 4];               // second run: K + 16
            const float4 c3 = x4[q + 5];
            afrag[kk][0]  = (_Float16)c0.x;  afrag[kk][1]  = (_Float16)c0.y;
            afrag[kk][2]  = (_Float16)c0.z;  afrag[kk][3]  = (_Float16)c0.w;
            afrag[kk][4]  = (_Float16)c1.x;  afrag[kk][5]  = (_Float16)c1.y;
            afrag[kk][6]  = (_Float16)c1.z;  afrag[kk][7]  = (_Float16)c1.w;
            afrag[kk][8]  = (_Float16)c2.x;  afrag[kk][9]  = (_Float16)c2.y;
            afrag[kk][10] = (_Float16)c2.z;  afrag[kk][11] = (_Float16)c2.w;
            afrag[kk][12] = (_Float16)c3.x;  afrag[kk][13] = (_Float16)c3.y;
            afrag[kk][14] = (_Float16)c3.z;  afrag[kk][15] = (_Float16)c3.w;
        }
    }

    // Each wave produces 4 of the 32 column tiles; B fragments come pre-swizzled
    // from d_ws as one contiguous 32B per-lane load (coalesced b128 pairs).
#pragma unroll
    for (int t = 0; t < 4; ++t) {
        const int tile = wave * 4 + t;     // 0..31
        const int col  = tile * 16 + r;    // 0..511
        v8f acc = {};
#pragma unroll
        for (int kk = 0; kk < 2; ++kk) {
            const v16h bfrag =
                *(const v16h*)(wsT + ((size_t)((tile * 2 + kk) * 32 + lane) << 4));
            acc = __builtin_amdgcn_wmma_f32_16x16x32_f16(
                /*neg_a=*/false, afrag[kk], /*neg_b=*/false, bfrag,
                /*c_mod=*/(short)0, acc, /*reuse_a=*/false, /*reuse_b=*/false);
        }
        // C/D f32 layout: vgpr v holds row M = v + 8*half, column = r.
#pragma unroll
        for (int v = 0; v < 8; ++v) {
            Mtile[(v + 8 * half) * KD + col] = acc[v];
        }
    }

    // x passthrough, float4-vectorized (row stride 96 floats keeps 16B alignment).
    {
        const int rr = tid >> 4;            // 0..15
        const int f4 = tid & 15;            // 0..15 -> covers 64 floats
        const float4* xs = (const float4*)(x + (size_t)(n0 + rr) * F_DIM);
        float4* od = (float4*)(out + (size_t)(n0 + rr) * OUT_F);
        od[f4] = xs[f4];
    }

    __syncthreads();

    // ---- Phase 2: feats[n,k] = sum_d exp(-sum_j |M[n,j,d]-M[n,k,d]|) ----
    // 512 (n,k) pairs, 2 per thread; all lanes of a wave share n -> LDS broadcast.
#pragma unroll
    for (int pp = 0; pp < 2; ++pp) {
        const int p = tid + pp * THREADS;   // 0..511
        const int n = p >> 5;               // row within tile
        const int k = p & 31;               // kernel index
        const float* Mrow = &Mtile[n * KD];

        float mk[D_KER];
#pragma unroll
        for (int d = 0; d < D_KER; ++d) mk[d] = Mrow[k * D_KER + d];

        float a[D_KER];
#pragma unroll
        for (int d = 0; d < D_KER; ++d) a[d] = 0.0f;

        for (int j = 0; j < K_KER; ++j) {
            const float4* mj = (const float4*)(Mrow + j * D_KER);
            const float4 q0 = mj[0], q1 = mj[1], q2 = mj[2], q3 = mj[3];
            a[0]  += fabsf(q0.x - mk[0]);   a[1]  += fabsf(q0.y - mk[1]);
            a[2]  += fabsf(q0.z - mk[2]);   a[3]  += fabsf(q0.w - mk[3]);
            a[4]  += fabsf(q1.x - mk[4]);   a[5]  += fabsf(q1.y - mk[5]);
            a[6]  += fabsf(q1.z - mk[6]);   a[7]  += fabsf(q1.w - mk[7]);
            a[8]  += fabsf(q2.x - mk[8]);   a[9]  += fabsf(q2.y - mk[9]);
            a[10] += fabsf(q2.z - mk[10]);  a[11] += fabsf(q2.w - mk[11]);
            a[12] += fabsf(q3.x - mk[12]);  a[13] += fabsf(q3.y - mk[13]);
            a[14] += fabsf(q3.z - mk[14]);  a[15] += fabsf(q3.w - mk[15]);
        }

        float acc = 0.0f;
#pragma unroll
        for (int d = 0; d < D_KER; ++d) acc += __expf(-a[d]);

        out[(size_t)(n0 + n) * OUT_F + F_DIM + k] = acc;
    }
}

extern "C" void kernel_launch(void* const* d_in, const int* in_sizes, int n_in,
                              void* d_out, int out_size, void* d_ws, size_t ws_size,
                              hipStream_t stream) {
    const float* x = (const float*)d_in[0];   // (B,S,F)   = 262144 floats
    const float* T = (const float*)d_in[1];   // (F,K*D)   = 32768 floats
    float* out = (float*)d_out;               // (B,S,F+K) = 393216 floats
    _Float16* wsT = (_Float16*)d_ws;          // 64 KB fragment-swizzled f16 T
    (void)in_sizes; (void)n_in; (void)ws_size; (void)out_size;

    // One-time (per call) fragment-swizzle + f16 convert of T: 32768 elems.
    prep_T_kernel<<<T_FRAG_ELEMS / THREADS, THREADS, 0, stream>>>(T, wsT);

    // Fused GEMM + minibatch-discrimination: 256 blocks x 256 threads (8 wave32s).
    mbd_fused_kernel<<<N_TOTAL / ROWS_PER_BLOCK, THREADS, 0, stream>>>(x, wsT, out);
}
